// MoEAttention_67130338836940
// MI455X (gfx1250) — compile-verified, matrix operand order checked by reading, not run on hardware
//
#include <hip/hip_runtime.h>

typedef __attribute__((ext_vector_type(16))) __bf16 v16bf;
typedef __attribute__((ext_vector_type(8)))  float  v8f;
typedef int v4i32 __attribute__((vector_size(16)));
#if defined(__AMDGCN__)
typedef v4i32 __attribute__((address_space(1))) as1_v4i;
typedef v4i32 __attribute__((address_space(3))) as3_v4i;
#endif

#define NTOK 2048
#define HDIM 4096
#define NHEAD 32
#define NKVH 8
#define HEADD 128
#define QKVW 6144
#define NEXP 8
#define SCALE 0.08838834764831845f  // 1/sqrt(128)

#if defined(__AMDGCN__) && __has_builtin(__builtin_amdgcn_global_load_async_to_lds_b128)
#define HAVE_ASYNC_LDS 1
#else
#define HAVE_ASYNC_LDS 0
#endif

__device__ inline void wait_async0() {
#if defined(__AMDGCN__)
#if __has_builtin(__builtin_amdgcn_s_wait_asynccnt)
    __builtin_amdgcn_s_wait_asynccnt(0);
#else
    asm volatile("s_wait_asynccnt 0x0" ::: "memory");
#endif
#endif
}

// ---------------------------------------------------------------------------
// Fragment load: 16x32 bf16 tile, row-major in LDS with `stride` elements.
// Matches CDNA5 16-bit A-matrix layout: lane L (row = L&15, half h = L>>4)
// holds K = h*8+0..7 in elements 0..7 and K = 16+h*8+0..7 in elements 8..15.
// B-fragments use the same pattern when the tile is stored column-major
// (i.e. tile[n][k] with k contiguous).
// ---------------------------------------------------------------------------
__device__ inline v16bf frag_ld(const __bf16* base, int stride) {
    const int lane = threadIdx.x & 31;
    const __bf16* p = base + (lane & 15) * stride + (lane >> 4) * 8;
    v16bf f;
    *((uint4*)&f)     = *(const uint4*)p;
    *((uint4*)&f + 1) = *(const uint4*)(p + 16);
    return f;
}

__device__ inline v8f wmma_bf16(v16bf a, v16bf b, v8f c) {
    return __builtin_amdgcn_wmma_f32_16x16x32_bf16(false, a, false, b,
                                                   (short)0, c, false, false);
}

__device__ inline unsigned pack_bf16(float lo, float hi) {
    union { unsigned u; __bf16 h[2]; } pk;
    pk.h[0] = (__bf16)lo;
    pk.h[1] = (__bf16)hi;
    return pk.u;
}

// ---------------------------------------------------------------------------
// RMSNorm -> bf16 (pre-attention)
// ---------------------------------------------------------------------------
__global__ __launch_bounds__(256) void k_rms_pre(const float* __restrict__ x,
                                                 const float* __restrict__ w,
                                                 __bf16* __restrict__ out) {
    __shared__ float red[8];
    __shared__ float tot;
    const long row = blockIdx.x;
    const float* xr = x + row * HDIM;
    float ss = 0.f;
    for (int i = threadIdx.x; i < HDIM; i += 256) { float v = xr[i]; ss += v * v; }
#pragma unroll
    for (int o = 16; o > 0; o >>= 1) ss += __shfl_xor(ss, o, 32);
    if ((threadIdx.x & 31) == 0) red[threadIdx.x >> 5] = ss;
    __syncthreads();
    if (threadIdx.x == 0) {
        float t = 0.f;
#pragma unroll
        for (int i = 0; i < 8; ++i) t += red[i];
        tot = t;
    }
    __syncthreads();
    const float sc = rsqrtf(tot / (float)HDIM + 1e-6f);
    for (int i = threadIdx.x; i < HDIM; i += 256)
        out[row * HDIM + i] = (__bf16)(xr[i] * sc * w[i]);
}

// ---------------------------------------------------------------------------
// RMSNorm -> f32 (post-attention)
// ---------------------------------------------------------------------------
__global__ __launch_bounds__(256) void k_rms_post(const float* __restrict__ x,
                                                  const float* __restrict__ w,
                                                  float* __restrict__ out) {
    __shared__ float red[8];
    __shared__ float tot;
    const long row = blockIdx.x;
    const float* xr = x + row * HDIM;
    float ss = 0.f;
    for (int i = threadIdx.x; i < HDIM; i += 256) { float v = xr[i]; ss += v * v; }
#pragma unroll
    for (int o = 16; o > 0; o >>= 1) ss += __shfl_xor(ss, o, 32);
    if ((threadIdx.x & 31) == 0) red[threadIdx.x >> 5] = ss;
    __syncthreads();
    if (threadIdx.x == 0) {
        float t = 0.f;
#pragma unroll
        for (int i = 0; i < 8; ++i) t += red[i];
        tot = t;
    }
    __syncthreads();
    const float sc = rsqrtf(tot / (float)HDIM + 1e-6f);
    for (int i = threadIdx.x; i < HDIM; i += 256)
        out[row * HDIM + i] = xr[i] * sc * w[i];
}

// ---------------------------------------------------------------------------
// WMMA GEMM: C[M x Nn] = A(bf16, MxK row-major) @ B(f32, KxNn row-major)
// Optional fused residual add (compile-time). grid = (M/128, Nn/128),
// block = 256 (8 waves); each wave computes a 32x64 sub-tile.
// Double-buffered LDS, async global->LDS copies for the A tile (ASYNCcnt),
// synchronous convert+transpose staging for the B tile, prefetch 2 ahead.
// ---------------------------------------------------------------------------
template <bool RESID>
__global__ __launch_bounds__(256) void k_gemm(const __bf16* __restrict__ A,
                                              const float* __restrict__ B,
                                              const float* __restrict__ resid,
                                              float* __restrict__ C,
                                              int K, int Nn) {
    __shared__ __bf16 As[2][128][40];   // [m][k] row-major
    __shared__ __bf16 Bs[2][128][40];   // [n][k] (transposed from global)
    const int tid  = threadIdx.x;
    const int wave = tid >> 5, lane = tid & 31;
    const int m_off = (wave >> 1) * 32, n_off = (wave & 1) * 64;
    const long row0 = (long)blockIdx.x * 128;
    const long col0 = (long)blockIdx.y * 128;

    const v8f vz = {0.f, 0.f, 0.f, 0.f, 0.f, 0.f, 0.f, 0.f};
    v8f acc[2][4];
#pragma unroll
    for (int i = 0; i < 2; ++i)
#pragma unroll
        for (int j = 0; j < 4; ++j) acc[i][j] = vz;

    const int ar = tid >> 1,  ac = (tid & 1) * 16;       // A staging coords
    const int bn = (tid & 15) * 8, bk = (tid >> 4) * 2;  // B staging: 8n x 2k

    auto stage = [&](int buf, int kc) {
        const __bf16* ga = A + (row0 + ar) * (long)K + kc + ac;
#if HAVE_ASYNC_LDS
        __builtin_amdgcn_global_load_async_to_lds_b128(
            (as1_v4i*)(void*)ga,
            (as3_v4i*)(void*)&As[buf][ar][ac],
            0, 0);
        __builtin_amdgcn_global_load_async_to_lds_b128(
            (as1_v4i*)(void*)(ga + 8),
            (as3_v4i*)(void*)&As[buf][ar][ac + 8],
            0, 0);
#else
        *(uint4*)&As[buf][ar][ac]     = *(const uint4*)ga;
        *(uint4*)&As[buf][ar][ac + 8] = *(const uint4*)(ga + 8);
#endif
        const float* gb0 = B + (long)(kc + bk) * Nn + col0 + bn;
        const float* gb1 = gb0 + Nn;
#pragma unroll
        for (int i = 0; i < 8; ++i)
            *(unsigned*)&Bs[buf][bn + i][bk] = pack_bf16(gb0[i], gb1[i]);
    };

    stage(0, 0);
    int buf = 0;
    for (int kc = 0; kc < K; kc += 32) {
#if HAVE_ASYNC_LDS
        wait_async0();   // A tile staged last iteration is now resident
#endif
        __syncthreads();
        if (kc + 32 < K) stage(buf ^ 1, kc + 32);
        if (kc + 64 < K) {
            __builtin_prefetch(A + (row0 + ar) * (long)K + kc + 64 + ac, 0, 1);
            __builtin_prefetch(B + (long)(kc + 64 + bk) * Nn + col0 + bn, 0, 1);
        }
        v16bf af0 = frag_ld(&As[buf][m_off][0], 40);
        v16bf af1 = frag_ld(&As[buf][m_off + 16][0], 40);
#pragma unroll
        for (int j = 0; j < 4; ++j) {
            v16bf bfj = frag_ld(&Bs[buf][n_off + 16 * j][0], 40);
            acc[0][j] = wmma_bf16(af0, bfj, acc[0][j]);
            acc[1][j] = wmma_bf16(af1, bfj, acc[1][j]);
        }
        buf ^= 1;
    }

    const int nl = lane & 15, hh = lane >> 4;
#pragma unroll
    for (int i = 0; i < 2; ++i)
#pragma unroll
        for (int j = 0; j < 4; ++j)
#pragma unroll
            for (int r = 0; r < 8; ++r) {
                const long m = row0 + m_off + i * 16 + hh * 8 + r;
                const long n = col0 + n_off + j * 16 + nl;
                float v = acc[i][j][r];
                if (RESID) v += resid[m * Nn + n];
                C[m * Nn + n] = v;
            }
}

// ---------------------------------------------------------------------------
// NeoX RoPE in place on q (32 heads) and k (8 heads) within qkv[2048][6144].
// grid = (2048, 40), block = 64 (thread i handles rotation pair (i, i+64)).
// ---------------------------------------------------------------------------
__global__ __launch_bounds__(64) void k_rope(float* __restrict__ qkv,
                                             const int* __restrict__ pos) {
    const int t = blockIdx.x, hidx = blockIdx.y, i = threadIdx.x;
    const long off = (long)t * QKVW +
                     ((hidx < NHEAD) ? hidx * HEADD
                                     : NHEAD * HEADD + (hidx - NHEAD) * HEADD);
    float* p = qkv + off;
    const float fp = (float)pos[t];
    // inv_freq = 10000^(-2i/128) = exp(-2i/128 * ln(10000))
    const float inv = __expf(-(float)(2 * i) * (9.210340371976184f / 128.0f));
    const float ang = fp * inv;
    float s, c;
    __sincosf(ang, &s, &c);
    const float x1 = p[i], x2 = p[i + 64];
    p[i]      = x1 * c - x2 * s;
    p[i + 64] = x2 * c + x1 * s;
}

// ---------------------------------------------------------------------------
// Flash-style causal GQA attention. grid = (16 q-blocks, 32 heads),
// block = 256 = 8 waves; wave w owns q rows [q0 + 16w, q0 + 16w + 15].
// K/V double-buffered in LDS (V transposed); QK^T and P*V via WMMA.
// Softmax is fully register-resident: the WMMA D-layout replicates each row
// across the 16 lanes of a half, so row max/sum are 4 xor-shuffles and the
// per-row correction factor lands directly in accumulator layout.
// Only the bf16 P tile round-trips LDS (mandatory D->A relayout).
// ---------------------------------------------------------------------------
__global__ __launch_bounds__(256) void k_attn(const float* __restrict__ qkv,
                                              __bf16* __restrict__ attn_out) {
    __shared__ __bf16 Kt[2][32][136];    // [key][d]
    __shared__ __bf16 Vt[2][HEADD][40];  // [d][key]  (transposed)
    __shared__ __bf16 Pt[8][16][32];     // per-wave P tile (A-layout friendly)

    const int tid = threadIdx.x, wave = tid >> 5, lane = tid & 31;
    const int head = blockIdx.y;
    const int kvh  = head >> 2;                    // GQA group of 4
    const int q0   = blockIdx.x * 128;
    const int qrow_base = q0 + wave * 16;
    const int nl = lane & 15, hh = lane >> 4;

    // Load Q fragments once (RoPE already applied), convert f32 -> bf16.
    v16bf qf[4];
    {
        const float* qp = qkv + (long)(qrow_base + nl) * QKVW + head * HEADD;
#pragma unroll
        for (int c = 0; c < 4; ++c) {
#pragma unroll
            for (int i = 0; i < 8; ++i) {
                qf[c][i]     = (__bf16)qp[c * 32 + hh * 8 + i];
                qf[c][8 + i] = (__bf16)qp[c * 32 + 16 + hh * 8 + i];
            }
        }
    }

    // Staging coordinates
    const int skey = tid >> 3, sds = (tid & 7) * 16;        // K: 1 key x 16 d
    const int vkey = (tid & 15) * 2, vds = (tid >> 4) * 8;  // V: 2 keys x 8 d

    auto stageKV = [&](int buf, int kstart) {
        // K tile: row-major [key][d], two b128 stores per thread.
        const float* kp = qkv + (long)(kstart + skey) * QKVW +
                          NHEAD * HEADD + kvh * HEADD + sds;
        __bf16 tmp[16];
#pragma unroll
        for (int i = 0; i < 16; ++i) tmp[i] = (__bf16)kp[i];
        *(uint4*)&Kt[buf][skey][sds]     = *(uint4*)&tmp[0];
        *(uint4*)&Kt[buf][skey][sds + 8] = *(uint4*)&tmp[8];
        // V tile transposed [d][key], packed b32 stores (key pair per word).
        const float* vp0 = qkv + (long)(kstart + vkey) * QKVW +
                           (NHEAD + NKVH) * HEADD + kvh * HEADD + vds;
        const float* vp1 = vp0 + QKVW;
#pragma unroll
        for (int i = 0; i < 8; ++i)
            *(unsigned*)&Vt[buf][vds + i][vkey] = pack_bf16(vp0[i], vp1[i]);
    };

    const v8f vz = {0.f, 0.f, 0.f, 0.f, 0.f, 0.f, 0.f, 0.f};
    v8f oacc[8];
#pragma unroll
    for (int j = 0; j < 8; ++j) oacc[j] = vz;
    float run_max[8], run_sum[8];
#pragma unroll
    for (int r = 0; r < 8; ++r) { run_max[r] = -3e38f; run_sum[r] = 0.f; }

    const int nkb = (q0 >> 5) + 4;   // covers keys 0 .. q0+127
    stageKV(0, 0);
    int buf = 0;
    for (int b = 0; b < nkb; ++b) {
        const int kstart = b * 32;
        __syncthreads();
        if (b + 1 < nkb) stageKV(buf ^ 1, kstart + 32);

        // S = Q @ K^T for 32 keys (two 16x16 D tiles, kept in registers).
        v8f sfr[2];
#pragma unroll
        for (int t = 0; t < 2; ++t) {
            v8f s = vz;
#pragma unroll
            for (int c = 0; c < 4; ++c) {
                v16bf kf = frag_ld(&Kt[buf][t * 16][c * 32], 136);
                s = wmma_bf16(qf[c], kf, s);
            }
            sfr[t] = s;
        }

        // Register-resident online softmax. Lane holds row m = hh*8+r at
        // key columns kstart+nl (t=0) and kstart+16+nl (t=1).
#pragma unroll
        for (int r = 0; r < 8; ++r) {
            const int qrow = qrow_base + hh * 8 + r;
            float x0 = sfr[0][r] * SCALE;
            float x1 = sfr[1][r] * SCALE;
            if (kstart + nl > qrow)      x0 = -3e38f;
            if (kstart + 16 + nl > qrow) x1 = -3e38f;
            float bm = fmaxf(x0, x1);
            bm = fmaxf(bm, __shfl_xor(bm, 1, 32));
            bm = fmaxf(bm, __shfl_xor(bm, 2, 32));
            bm = fmaxf(bm, __shfl_xor(bm, 4, 32));
            bm = fmaxf(bm, __shfl_xor(bm, 8, 32));
            const float nmax = fmaxf(run_max[r], bm);
            const float corr = __expf(run_max[r] - nmax);
            run_max[r] = nmax;
            const float p0 = __expf(x0 - nmax);
            const float p1 = __expf(x1 - nmax);
            float ps = p0 + p1;
            ps += __shfl_xor(ps, 1, 32);
            ps += __shfl_xor(ps, 2, 32);
            ps += __shfl_xor(ps, 4, 32);
            ps += __shfl_xor(ps, 8, 32);
            run_sum[r] = run_sum[r] * corr + ps;
            Pt[wave][hh * 8 + r][nl]      = (__bf16)p0;
            Pt[wave][hh * 8 + r][16 + nl] = (__bf16)p1;
            // Rescale accumulator row r (corr already in D-layout).
#pragma unroll
            for (int j = 0; j < 8; ++j) oacc[j][r] *= corr;
        }

        // O += P @ V  (P relayout via LDS, same-wave DS ordering).
        v16bf pf = frag_ld(&Pt[wave][0][0], 32);
#pragma unroll
        for (int j = 0; j < 8; ++j) {
            v16bf vf = frag_ld(&Vt[buf][j * 16][0], 40);
            oacc[j] = wmma_bf16(pf, vf, oacc[j]);
        }
        buf ^= 1;
    }

    float invl[8];
#pragma unroll
    for (int r = 0; r < 8; ++r) invl[r] = 1.0f / run_sum[r];
#pragma unroll
    for (int j = 0; j < 8; ++j)
#pragma unroll
        for (int r = 0; r < 8; ++r) {
            const long row = qrow_base + hh * 8 + r;
            attn_out[row * HDIM + head * HEADD + j * 16 + nl] =
                (__bf16)(oacc[j][r] * invl[r]);
        }
}

// ---------------------------------------------------------------------------
// Top-k (k=2 of 8) + normalized weights + stable counting-sort argsort.
// Single block, 256 threads (8 waves; wave e handles expert e in the sort).
// ---------------------------------------------------------------------------
__global__ __launch_bounds__(256) void k_topk(const float* __restrict__ logits,
                                              float* __restrict__ tw,
                                              int* __restrict__ tids,
                                              int* __restrict__ reorder) {
    __shared__ int flat[NTOK * 2];
    __shared__ int cnts[NEXP];
    const int tid = threadIdx.x;

    for (int t = tid; t < NTOK; t += 256) {
        const float* l = logits + t * NEXP;
        float b1 = -3e38f; int i1 = 0;
#pragma unroll
        for (int e = 0; e < NEXP; ++e)
            if (l[e] > b1) { b1 = l[e]; i1 = e; }
        float b2 = -3e38f; int i2 = 0;
#pragma unroll
        for (int e = 0; e < NEXP; ++e)
            if (e != i1 && l[e] > b2) { b2 = l[e]; i2 = e; }
        const float s = b1 + b2;
        tw[t * 2]     = b1 / s;
        tw[t * 2 + 1] = b2 / s;
        tids[t * 2]     = i1;
        tids[t * 2 + 1] = i2;
        flat[t * 2]     = i1;
        flat[t * 2 + 1] = i2;
    }
    __syncthreads();

    const int wave = tid >> 5, lane = tid & 31;
    // Pass 1: count occurrences of expert `wave`.
    int cnt = 0;
    for (int base = 0; base < NTOK * 2; base += 32)
        cnt += (flat[base + lane] == wave) ? 1 : 0;
#pragma unroll
    for (int o = 16; o > 0; o >>= 1) cnt += __shfl_xor(cnt, o, 32);
    if (lane == 0) cnts[wave] = cnt;
    __syncthreads();

    int basepos = 0;
    for (int e = 0; e < wave; ++e) basepos += cnts[e];
    // Pass 2: stable placement via ballots.
    for (int base = 0; base < NTOK * 2; base += 32) {
        const bool p = (flat[base + lane] == wave);
        const unsigned m = (unsigned)__ballot(p);
        const unsigned lower = m & ((lane == 0) ? 0u : (0xFFFFFFFFu >> (32 - lane)));
        if (p) reorder[basepos + __popc(lower)] = base + lane;
        basepos += __popc(m);
    }
}

// ---------------------------------------------------------------------------
// permuted_output[r] = norm_out[reorder[r] / 2]
// ---------------------------------------------------------------------------
__global__ __launch_bounds__(256) void k_permute(const float* __restrict__ src,
                                                 const int* __restrict__ reorder,
                                                 float* __restrict__ dst) {
    const int r = blockIdx.x;
    const long s = (long)(reorder[r] >> 1);
    const float* sp = src + s * HDIM;
    float* dp = dst + (long)r * HDIM;
    for (int i = threadIdx.x; i < HDIM; i += 256) dp[i] = sp[i];
}

// ---------------------------------------------------------------------------
extern "C" void kernel_launch(void* const* d_in, const int* in_sizes, int n_in,
                              void* d_out, int out_size, void* d_ws, size_t ws_size,
                              hipStream_t stream) {
    (void)in_sizes; (void)n_in; (void)out_size; (void)ws_size;
    const int*   positions = (const int*)d_in[0];
    const float* hidden    = (const float*)d_in[1];
    const float* w_qkv     = (const float*)d_in[2];
    const float* w_o       = (const float*)d_in[3];
    // d_in[4] = w_gate : dead in the reference (router_logits unused)
    const float* w_pre     = (const float*)d_in[5];
    const float* w_post    = (const float*)d_in[6];
    const float* elogits   = (const float*)d_in[7];

    char* ws = (char*)d_ws;
    // Region A [0, 16MB): x_bf16, later reused as attn_bf16
    __bf16* x_bf    = (__bf16*)(ws);
    __bf16* attn_bf = (__bf16*)(ws);
    // Region B [16MB, 64MB): qkv f32 (48MB), later reused for post-norm f32 (32MB)
    float* qkv   = (float*)(ws + (16l << 20));
    float* normb = (float*)(ws + (16l << 20));
    // Region C [64MB, 96MB): s = o_proj + residual (32MB)
    float* s_buf = (float*)(ws + (64l << 20));

    float* perm_out = (float*)d_out;
    float* tw_out   = perm_out + (size_t)(NTOK * 2) * HDIM;   // 4096*4096
    int*   tids_out = (int*)(tw_out + NTOK * 2);
    int*   reo_out  = (int*)(tw_out + NTOK * 4);

    // 1) pre-RMSNorm -> bf16
    k_rms_pre<<<NTOK, 256, 0, stream>>>(hidden, w_pre, x_bf);
    // 2) QKV GEMM (2048x4096 @ 4096x6144) -> f32
    k_gemm<false><<<dim3(NTOK / 128, QKVW / 128), 256, 0, stream>>>(
        x_bf, w_qkv, nullptr, qkv, HDIM, QKVW);
    // 3) RoPE on q and k (40 heads)
    k_rope<<<dim3(NTOK, NHEAD + NKVH), 64, 0, stream>>>(qkv, positions);
    // 4) causal GQA flash attention -> bf16 (overwrites x_bf region)
    k_attn<<<dim3(NTOK / 128, NHEAD), 256, 0, stream>>>(qkv, attn_bf);
    // 5) O projection + residual -> s (f32)
    k_gemm<true><<<dim3(NTOK / 128, HDIM / 128), 256, 0, stream>>>(
        attn_bf, w_o, hidden, s_buf, HDIM, HDIM);
    // 6) post-RMSNorm -> f32 (overwrites qkv region)
    k_rms_post<<<NTOK, 256, 0, stream>>>(s_buf, w_post, normb);
    // 7) top-k weights/ids + stable argsort of flattened ids
    k_topk<<<1, 256, 0, stream>>>(elogits, tw_out, tids_out, reo_out);
    // 8) gather permuted rows
    k_permute<<<NTOK * 2, 256, 0, stream>>>(normb, reo_out, perm_out);
}
